// MBDCompressor3D_44702019616945
// MI455X (gfx1250) — matrix operand order repaired and checked by reading, not compiled, not run on hardware
//
#include <hip/hip_runtime.h>
#include <hip/hip_bf16.h>

// MBDCompressor3D for MI455X (gfx1250, wave32).
// 128 threads (4 waves) per block, 128 queries per block.
// GEMMs via v_wmma_f32_16x16x32_f16; fragments loaded as 2x ds_load_b128;
// coarse contraction fused into GEMM1 epilogue via ds_add_f32.
// Scalar sections use float4 / half8 LDS vector accesses; all output streams
// use non-temporal stores (outputs are 512MB streaming, > L2).

typedef _Float16 half16 __attribute__((ext_vector_type(16)));
typedef _Float16 half8  __attribute__((ext_vector_type(8)));
typedef float    float8 __attribute__((ext_vector_type(8)));
typedef float    float4v __attribute__((ext_vector_type(4)));

#define TPB 128
#define QPB 128

struct __align__(16) Smem {
  _Float16 arena1[QPB * 48];   // psiA (stride 32) -> h1 (stride 48)
  _Float16 arena2[QPB * 48];   // peA (stride 32, zero-padded) -> h2 (stride 48)
  _Float16 Bt[96 * 32];        // B transposed: [n=96][k=32]
  _Float16 fw0t[48 * 32];      // fw0 transposed: [n=48][k=32], k 27..31 zero
  _Float16 fw1t[48 * 48];      // [n=48][k=48]
  _Float16 fw2t[12 * 48];      // [n=12][k=48]
  float Pbuf[64 * 12];         // per-Gaussian: P00,2P01,2P02,P11,2P12,P22,mu(3),pad(3)
  float Cw[32 * 8];
  float fb0[48], fb1[48], fb2[12];
  float gw0[3 * 24], gb0[24], gw1[24], gb1[4];
  float rw0[15 * 24], rb0[24], rw1[24 * 12], rb1[12];
  float mcL[QPB * 8];
  float coarseL[QPB * 12];
  float fineL[QPB * 12];
};

__device__ inline float8 wmma16(half16 a, half16 b, float8 c) {
  return __builtin_amdgcn_wmma_f32_16x16x32_f16(false, a, false, b, (short)0, c,
                                                false, false);
}

// A fragment (16x32 MxK) from row-major f16 act[row][k], row stride ld (mult
// of 8). Lane layout (ISA 7.12.2): lane&15 = row; lane>>4 selects K-phase.
// Per lane: two contiguous 8-half runs at ko and ko+16 -> 2x ds_load_b128.
__device__ inline half16 load_a8(const _Float16* act, int m0, int ld,
                                 int kbase, int kmax, int lane) {
  int ko = kbase + ((lane >> 4) << 3);
  const _Float16* p = act + (m0 + (lane & 15)) * ld + ko;
  half8 z8 = {};
  half8 r0 = *(const half8*)p;
  half8 r1 = *(const half8*)(p + 16);
  if (ko + 16 >= kmax) r1 = z8;
  return __builtin_shufflevector(r0, r1, 0, 1, 2, 3, 4, 5, 6, 7, 8, 9, 10, 11,
                                 12, 13, 14, 15);
}

// B fragment (32x16 KxN) from TRANSPOSED f16 weights wt[n][k], k stride ldk
// (mult of 8). Lane: n = n0+(lane&15); two 8-half k runs at ko and ko+16.
__device__ inline half16 load_bt(const _Float16* wt, int n0, int nmax, int ldk,
                                 int kbase, int kmax, int lane) {
  int col = n0 + (lane & 15);
  int ko = kbase + ((lane >> 4) << 3);
  bool ok = (col < nmax) && (ko < kmax);
  const _Float16* p = wt + (ok ? (col * ldk + ko) : 0);
  half8 r0 = *(const half8*)p;
  half8 r1 = *(const half8*)(p + 16);
  if (ko + 16 >= kmax) r1 = (half8){};
  half16 b = __builtin_shufflevector(r0, r1, 0, 1, 2, 3, 4, 5, 6, 7, 8, 9, 10,
                                     11, 12, 13, 14, 15);
  half16 z = {};
  return ok ? b : z;
}

__device__ inline float4v vmax0(float4v v) {
#pragma unroll
  for (int u = 0; u < 4; ++u) v[u] = fmaxf(v[u], 0.f);
  return v;
}

__global__ __launch_bounds__(TPB) void mbd_kernel(
    const float* __restrict__ coords,
    const float* __restrict__ coeff_mu, const float* __restrict__ coeff_log_s,
    const float* __restrict__ coeff_q,
    const float* __restrict__ basis_mu, const float* __restrict__ basis_log_s,
    const float* __restrict__ basis_q,
    const float* __restrict__ Csrc, const float* __restrict__ Bsrc,
    const float* __restrict__ fw0p, const float* __restrict__ fb0p,
    const float* __restrict__ fw1p, const float* __restrict__ fb1p,
    const float* __restrict__ fw2p, const float* __restrict__ fb2p,
    const float* __restrict__ gw0p, const float* __restrict__ gb0p,
    const float* __restrict__ gw1p, const float* __restrict__ gb1p,
    const float* __restrict__ rw0p, const float* __restrict__ rb0p,
    const float* __restrict__ rw1p, const float* __restrict__ rb1p,
    float* __restrict__ rec_out, float* __restrict__ mc_out,
    float* __restrict__ mb_out, float* __restrict__ coarse_out) {
  __shared__ Smem sm;
  const int tid  = threadIdx.x;
  const int lane = tid & 31;
  const int wave = tid >> 5;
  const int q0   = blockIdx.x * QPB;
  const int q    = q0 + tid;

  // ---------------- stage 0: cooperative loads + precompute ----------------
  for (int i = tid; i < 96 * 32; i += TPB) {
    int n = i >> 5, k = i & 31;
    sm.Bt[i] = (_Float16)Bsrc[k * 96 + n];
  }
  for (int i = tid; i < 48 * 32; i += TPB) {
    int n = i >> 5, k = i & 31;
    sm.fw0t[i] = (k < 27) ? (_Float16)fw0p[k * 48 + n] : (_Float16)0.0f;
  }
  for (int i = tid; i < 48 * 48; i += TPB) {
    int n = i / 48, k = i - 48 * n;
    sm.fw1t[i] = (_Float16)fw1p[k * 48 + n];
  }
  for (int i = tid; i < 12 * 48; i += TPB) {
    int n = i / 48, k = i - 48 * n;
    sm.fw2t[i] = (_Float16)fw2p[k * 12 + n];
  }
  for (int i = tid; i < 32 * 8;  i += TPB) sm.Cw[i]  = Csrc[i];
  for (int i = tid; i < 15 * 24; i += TPB) sm.rw0[i] = rw0p[i];
  for (int i = tid; i < 24 * 12; i += TPB) sm.rw1[i] = rw1p[i];
  for (int i = tid; i < QPB * 12; i += TPB) sm.coarseL[i] = 0.0f;
  if (tid < 48) { sm.fb0[tid] = fb0p[tid]; sm.fb1[tid] = fb1p[tid]; }
  if (tid < 12) { sm.fb2[tid] = fb2p[tid]; sm.rb1[tid] = rb1p[tid]; }
  if (tid < 72) sm.gw0[tid] = gw0p[tid];
  if (tid < 24) { sm.gb0[tid] = gb0p[tid]; sm.gw1[tid] = gw1p[tid]; sm.rb0[tid] = rb0p[tid]; }
  if (tid == 0) sm.gb1[0] = gb1p[0];

  if (tid < 64) {  // one (symmetric, compacted) precision matrix per thread
    int k = tid;
    const float* mu = (k < 32) ? (coeff_mu + k * 3) : (basis_mu + (k - 32) * 3);
    const float* ls = (k < 32) ? (coeff_log_s + k * 3) : (basis_log_s + (k - 32) * 3);
    const float* qq = (k < 32) ? (coeff_q + k * 4) : (basis_q + (k - 32) * 4);
    float qw = qq[0], qx = qq[1], qy = qq[2], qz = qq[3];
    float nrm = sqrtf(qw * qw + qx * qx + qy * qy + qz * qz) + 1e-8f;
    float w = qw / nrm, x = qx / nrm, y = qy / nrm, z = qz / nrm;
    float R[9] = {1.f - 2.f * (y * y + z * z), 2.f * (x * y - w * z), 2.f * (x * z + w * y),
                  2.f * (x * y + w * z), 1.f - 2.f * (x * x + z * z), 2.f * (y * z - w * x),
                  2.f * (x * z - w * y), 2.f * (y * z + w * x), 1.f - 2.f * (x * x + y * y)};
    float s0 = __expf(ls[0]), s1 = __expf(ls[1]), s2 = __expf(ls[2]);
    float i0 = 1.f / (s0 * s0 + 1e-8f), i1 = 1.f / (s1 * s1 + 1e-8f), i2 = 1.f / (s2 * s2 + 1e-8f);
    float P00 = R[0] * i0 * R[0] + R[1] * i1 * R[1] + R[2] * i2 * R[2];
    float P01 = R[0] * i0 * R[3] + R[1] * i1 * R[4] + R[2] * i2 * R[5];
    float P02 = R[0] * i0 * R[6] + R[1] * i1 * R[7] + R[2] * i2 * R[8];
    float P11 = R[3] * i0 * R[3] + R[4] * i1 * R[4] + R[5] * i2 * R[5];
    float P12 = R[3] * i0 * R[6] + R[4] * i1 * R[7] + R[5] * i2 * R[8];
    float P22 = R[6] * i0 * R[6] + R[7] * i1 * R[7] + R[8] * i2 * R[8];
    float* Pr = &sm.Pbuf[k * 12];
    float4v pa = {P00, 2.f * P01, 2.f * P02, P11};
    float4v pb = {2.f * P12, P22, mu[0], mu[1]};
    float4v pc = {mu[2], 0.f, 0.f, 0.f};
    ((float4v*)Pr)[0] = pa; ((float4v*)Pr)[1] = pb; ((float4v*)Pr)[2] = pc;
  }
  __syncthreads();

  // ---------------- stage 1: per-thread scalar math -------------------------
  const float px = coords[q * 3 + 0], py = coords[q * 3 + 1], pz = coords[q * 3 + 2];

  float g[32];
  float ssum = 1e-8f;
#pragma unroll
  for (int m = 0; m < 32; ++m) {
    const float4v* P4 = (const float4v*)(sm.Pbuf + m * 12);
    float4v pa = P4[0], pb = P4[1], pc = P4[2];
    float dx = px - pb[2], dy = py - pb[3], dz = pz - pc[0];
    float maha = dx * (pa[0] * dx + pa[1] * dy + pa[2] * dz) +
                 dy * (pa[3] * dy + pb[0] * dz) + pb[1] * (dz * dz);
    float gv = __expf(-0.5f * maha);
    g[m] = gv; ssum += gv;
  }
  float inv = 1.0f / ssum;
  {
    float4v acc0 = {}, acc1 = {};
#pragma unroll
    for (int m = 0; m < 32; ++m) {
      const float4v* c4 = (const float4v*)(sm.Cw + m * 8);
      float gm = g[m];
      acc0 += gm * c4[0];
      acc1 += gm * c4[1];
    }
    acc0 *= inv; acc1 *= inv;
    ((float4v*)(sm.mcL + tid * 8))[0] = acc0;
    ((float4v*)(sm.mcL + tid * 8))[1] = acc1;
    float4v* mo = (float4v*)(mc_out + (size_t)q * 8);
    __builtin_nontemporal_store(acc0, mo + 0);
    __builtin_nontemporal_store(acc1, mo + 1);
  }

  ssum = 1e-8f;
#pragma unroll
  for (int n = 0; n < 32; ++n) {
    const float4v* P4 = (const float4v*)(sm.Pbuf + (32 + n) * 12);
    float4v pa = P4[0], pb = P4[1], pc = P4[2];
    float dx = px - pb[2], dy = py - pb[3], dz = pz - pc[0];
    float maha = dx * (pa[0] * dx + pa[1] * dy + pa[2] * dz) +
                 dy * (pa[3] * dy + pb[0] * dz) + pb[1] * (dz * dz);
    float gv = __expf(-0.5f * maha);
    g[n] = gv; ssum += gv;
  }
  inv = 1.0f / ssum;
  {
    half8* ps = (half8*)&sm.arena1[tid * 32];
#pragma unroll
    for (int v = 0; v < 4; ++v) {
      half8 hv;
#pragma unroll
      for (int j = 0; j < 8; ++j) hv[j] = (_Float16)(g[v * 8 + j] * inv);
      ps[v] = hv;
    }
  }

  {  // positional encoding -> arena2, stride 32 (27 valid + 5 zeros)
    float pe[32];
    pe[0] = px; pe[1] = py; pe[2] = pz;
    float f = 3.14159265358979323846f;
#pragma unroll
    for (int i = 0; i < 4; ++i) {
      pe[3 + 6 * i + 0] = __sinf(f * px);
      pe[3 + 6 * i + 1] = __sinf(f * py);
      pe[3 + 6 * i + 2] = __sinf(f * pz);
      pe[3 + 6 * i + 3] = __cosf(f * px);
      pe[3 + 6 * i + 4] = __cosf(f * py);
      pe[3 + 6 * i + 5] = __cosf(f * pz);
      f *= 2.0f;
    }
#pragma unroll
    for (int k = 27; k < 32; ++k) pe[k] = 0.0f;
    half8* pp = (half8*)&sm.arena2[tid * 32];
#pragma unroll
    for (int v = 0; v < 4; ++v) {
      half8 hv;
#pragma unroll
      for (int j = 0; j < 8; ++j) hv[j] = (_Float16)pe[v * 8 + j];
      pp[v] = hv;
    }
  }

  float gate;
  {
    float4v hg[6];
#pragma unroll
    for (int t = 0; t < 6; ++t) {
      hg[t] = ((const float4v*)sm.gb0)[t];
      hg[t] += px * ((const float4v*)(sm.gw0 + 0))[t];
      hg[t] += py * ((const float4v*)(sm.gw0 + 24))[t];
      hg[t] += pz * ((const float4v*)(sm.gw0 + 48))[t];
    }
    float gacc = sm.gb1[0];
#pragma unroll
    for (int t = 0; t < 6; ++t) {
      float4v wv = ((const float4v*)sm.gw1)[t];
      float4v hv = vmax0(hg[t]);
      gacc += hv[0] * wv[0] + hv[1] * wv[1] + hv[2] * wv[2] + hv[3] * wv[3];
    }
    gate = 1.f / (1.f + __expf(-gacc));
  }
  __syncthreads();

  // ------------- stage 2: GEMM1  psi[128,32] @ B[32,96] ---------------------
  for (int t = wave; t < 8 * 6; t += 4) {
    int mt = t / 6, nt = t % 6;
    half16 a = load_a8(sm.arena1, mt * 16, 32, 0, 32, lane);
    half16 b = load_bt(sm.Bt, nt * 16, 96, 32, 0, 32, lane);
    float8 c = {};
    c = wmma16(a, b, c);
    int col = nt * 16 + (lane & 15);
    int l = col / 12, d = col - l * 12;
    int rb_ = mt * 16 + ((lane >> 4) << 3);
    float mcv[8];
#pragma unroll
    for (int i = 0; i < 8; ++i) mcv[i] = sm.mcL[(rb_ + i) * 8 + l];
#pragma unroll
    for (int i = 0; i < 8; ++i) {
      int ql = rb_ + i;
      float v = c[i];
      __builtin_nontemporal_store(v, mb_out + (size_t)(q0 + ql) * 96 + col);
      atomicAdd(&sm.coarseL[ql * 12 + d], mcv[i] * v);
    }
  }
  __syncthreads();

  // ------------- stage 3: mlp0  pe[128,27] @ fw0[27,48] + relu --------------
  for (int t = wave; t < 8 * 3; t += 4) {
    int mt = t / 3, nt = t % 3;
    half16 a = load_a8(sm.arena2, mt * 16, 32, 0, 32, lane);
    half16 b = load_bt(sm.fw0t, nt * 16, 48, 32, 0, 32, lane);
    float8 c = {};
    c = wmma16(a, b, c);
    int col = nt * 16 + (lane & 15);
    float bias = sm.fb0[col];
    int rb_ = mt * 16 + ((lane >> 4) << 3);
#pragma unroll
    for (int i = 0; i < 8; ++i)
      sm.arena1[(rb_ + i) * 48 + col] = (_Float16)fmaxf(c[i] + bias, 0.f);
  }
  __syncthreads();

  // ------------- stage 4: mlp1  h1[128,48] @ fw1[48,48] + relu --------------
  for (int t = wave; t < 8 * 3; t += 4) {
    int mt = t / 3, nt = t % 3;
    half16 a0 = load_a8(sm.arena1, mt * 16, 48, 0, 48, lane);
    half16 a1 = load_a8(sm.arena1, mt * 16, 48, 32, 48, lane);
    half16 b0 = load_bt(sm.fw1t, nt * 16, 48, 48, 0, 48, lane);
    half16 b1 = load_bt(sm.fw1t, nt * 16, 48, 48, 32, 48, lane);
    float8 c = {};
    c = wmma16(a0, b0, c);
    c = wmma16(a1, b1, c);
    int col = nt * 16 + (lane & 15);
    float bias = sm.fb1[col];
    int rb_ = mt * 16 + ((lane >> 4) << 3);
#pragma unroll
    for (int i = 0; i < 8; ++i)
      sm.arena2[(rb_ + i) * 48 + col] = (_Float16)fmaxf(c[i] + bias, 0.f);
  }
  __syncthreads();

  // ------------- stage 5: mlp2  h2[128,48] @ fw2[48,12] ---------------------
  for (int t = wave; t < 8; t += 4) {
    int mt = t;
    half16 a0 = load_a8(sm.arena2, mt * 16, 48, 0, 48, lane);
    half16 a1 = load_a8(sm.arena2, mt * 16, 48, 32, 48, lane);
    half16 b0 = load_bt(sm.fw2t, 0, 12, 48, 0, 48, lane);
    half16 b1 = load_bt(sm.fw2t, 0, 12, 48, 32, 48, lane);
    float8 c = {};
    c = wmma16(a0, b0, c);
    c = wmma16(a1, b1, c);
    int col = lane & 15;
    int rb_ = mt * 16 + ((lane >> 4) << 3);
    if (col < 12) {
      float bias = sm.fb2[col];
#pragma unroll
      for (int i = 0; i < 8; ++i)
        sm.fineL[(rb_ + i) * 12 + col] = c[i] + bias;
    }
  }
  __syncthreads();

  // ------------- stage 6: blend + residual refine + outputs -----------------
  float4v bl4[3];
  {
    float4v cv4[3];
#pragma unroll
    for (int t = 0; t < 3; ++t) {
      cv4[t] = ((const float4v*)(sm.coarseL + tid * 12))[t];
      float4v fn = ((const float4v*)(sm.fineL + tid * 12))[t];
      bl4[t] = (1.f - gate) * cv4[t] + gate * fn;
    }
    float4v* co = (float4v*)(coarse_out + (size_t)q * 12);
    __builtin_nontemporal_store(cv4[0], co + 0);
    __builtin_nontemporal_store(cv4[1], co + 1);
    __builtin_nontemporal_store(cv4[2], co + 2);
  }
  // residual refiner: in = [blended(12), coords(3)] -> 24 (relu) -> 12
  float4v hh[6];
#pragma unroll
  for (int t = 0; t < 6; ++t) hh[t] = ((const float4v*)sm.rb0)[t];
#pragma unroll
  for (int k = 0; k < 15; ++k) {
    float val = (k < 12) ? bl4[k >> 2][k & 3]
                         : ((k == 12) ? px : ((k == 13) ? py : pz));
    const float4v* w = (const float4v*)(sm.rw0 + k * 24);
#pragma unroll
    for (int t = 0; t < 6; ++t) hh[t] += val * w[t];
  }
  float4v r4[3];
#pragma unroll
  for (int t = 0; t < 3; ++t) r4[t] = ((const float4v*)sm.rb1)[t];
#pragma unroll
  for (int t = 0; t < 6; ++t) {
    float4v hv = vmax0(hh[t]);
#pragma unroll
    for (int u = 0; u < 4; ++u) {
      const float4v* w = (const float4v*)(sm.rw1 + (4 * t + u) * 12);
      r4[0] += hv[u] * w[0];
      r4[1] += hv[u] * w[1];
      r4[2] += hv[u] * w[2];
    }
  }
  {
    float4v* ro = (float4v*)(rec_out + (size_t)q * 12);
#pragma unroll
    for (int t = 0; t < 3; ++t) {
      float4v rv = bl4[t] + 0.1f * r4[t];
      __builtin_nontemporal_store(rv, ro + t);
    }
  }
}

extern "C" void kernel_launch(void* const* d_in, const int* in_sizes, int n_in,
                              void* d_out, int out_size, void* d_ws, size_t ws_size,
                              hipStream_t stream) {
  (void)n_in; (void)out_size; (void)d_ws; (void)ws_size;
  const float* coords      = (const float*)d_in[0];
  const float* coeff_mu    = (const float*)d_in[1];
  const float* coeff_log_s = (const float*)d_in[2];
  const float* coeff_q     = (const float*)d_in[3];
  const float* basis_mu    = (const float*)d_in[4];
  const float* basis_log_s = (const float*)d_in[5];
  const float* basis_q     = (const float*)d_in[6];
  const float* C           = (const float*)d_in[7];
  const float* B           = (const float*)d_in[8];
  const float* fw0 = (const float*)d_in[9];  const float* fb0 = (const float*)d_in[10];
  const float* fw1 = (const float*)d_in[11]; const float* fb1 = (const float*)d_in[12];
  const float* fw2 = (const float*)d_in[13]; const float* fb2 = (const float*)d_in[14];
  const float* gw0 = (const float*)d_in[15]; const float* gb0 = (const float*)d_in[16];
  const float* gw1 = (const float*)d_in[17]; const float* gb1 = (const float*)d_in[18];
  const float* rw0 = (const float*)d_in[19]; const float* rb0 = (const float*)d_in[20];
  const float* rw1 = (const float*)d_in[21]; const float* rb1 = (const float*)d_in[22];

  const int Q = in_sizes[0] / 3;
  float* out        = (float*)d_out;
  float* rec_out    = out;
  float* mc_out     = out + (size_t)Q * 12;
  float* mb_out     = out + (size_t)Q * 20;
  float* coarse_out = out + (size_t)Q * 116;

  dim3 grid(Q / QPB), block(TPB);
  mbd_kernel<<<grid, block, 0, stream>>>(
      coords, coeff_mu, coeff_log_s, coeff_q, basis_mu, basis_log_s, basis_q,
      C, B, fw0, fb0, fw1, fb1, fw2, fb2, gw0, gb0, gw1, gb1, rw0, rb0, rw1, rb1,
      rec_out, mc_out, mb_out, coarse_out);
}
